// ContrastiveLoss_for_TSformer_SA_1537598292252
// MI455X (gfx1250) — matrix operand order repaired and checked by reading, not compiled, not run on hardware
//
#include <hip/hip_runtime.h>

// ---------------------------------------------------------------------------
// NT-Xent contrastive loss, fused: normalize -> bf16 hi/lo split ->
// WMMA bf16 (3-term fp32-accurate split) tiled sim -> fused exp/rowsum ->
// deterministic scalar reduction.
// B=4096, D=256, 2B=8192 rows, temperature 0.5 (invT = 2).
//
// CDNA5-specific paths used:
//   * v_wmma_f32_16x16x32_bf16 (24 per 16x16 tile, 3-term bf16 split ~ fp32)
//   * global_load_async_to_lds_b128 (ASYNCcnt DMA, double-buffered B tiles)
//   * s_wait_asynccnt split-barrier pipeline
// ---------------------------------------------------------------------------

typedef __attribute__((ext_vector_type(16))) __bf16 v16bf;
typedef __attribute__((ext_vector_type(8)))  float  v8f;

union ABFrag {
    v16bf v;
    uint4 u[2];   // 32 bytes
};

#define ROWS2B 8192
#define DDIM   256
#define BHALF  4096
#define INV_T  2.0f
#define NTILES 512      // 8192 / 16 column tiles
#define KSTEPS 8        // 256 / 32

__device__ __forceinline__ unsigned short f32_to_bf16_rne(float f) {
    unsigned int u = __float_as_uint(f);
    u += 0x7FFFu + ((u >> 16) & 1u);       // round-to-nearest-even
    return (unsigned short)(u >> 16);
}
__device__ __forceinline__ float bf16_to_f32(unsigned short h) {
    return __uint_as_float(((unsigned int)h) << 16);
}

// Async DMA: global -> LDS, 16 bytes per lane, tracked by ASYNCcnt.
__device__ __forceinline__ void async_load_b128(unsigned lds_addr, const void* gptr) {
    asm volatile("global_load_async_to_lds_b128 %0, %1, off"
                 :: "v"(lds_addr), "v"((unsigned long long)(uintptr_t)gptr)
                 : "memory");
}

__device__ __forceinline__ void wait_async0() {
#if __has_builtin(__builtin_amdgcn_s_wait_asynccnt)
    __builtin_amdgcn_s_wait_asynccnt(0);
#else
    asm volatile("s_wait_asynccnt 0" ::: "memory");
#endif
}

// --------------------------------------------------------------------------
// Kernel 1: row L2-normalize (eps=1e-12, F.normalize semantics) and split the
// normalized fp32 value into bf16 hi + bf16 lo planes. One wave per row.
// --------------------------------------------------------------------------
__global__ __launch_bounds__(256)
void normalize_split_kernel(const float* __restrict__ emb_i,
                            const float* __restrict__ emb_j,
                            unsigned short* __restrict__ rep_hi,
                            unsigned short* __restrict__ rep_lo) {
    const int row  = blockIdx.x * 8 + (threadIdx.x >> 5);
    const int lane = threadIdx.x & 31;
    if (row >= ROWS2B) return;

    const float* src = (row < BHALF) ? (emb_i + (size_t)row * DDIM)
                                     : (emb_j + (size_t)(row - BHALF) * DDIM);
    float s = 0.0f;
    for (int j = lane; j < DDIM; j += 32) {
        float x = src[j];
        s += x * x;
    }
    for (int m = 1; m < 32; m <<= 1) s += __shfl_xor(s, m, 32);
    const float scale = 1.0f / fmaxf(sqrtf(s), 1e-12f);

    for (int j = lane; j < DDIM; j += 32) {
        float y = src[j] * scale;
        unsigned short h = f32_to_bf16_rne(y);
        float lo = y - bf16_to_f32(h);
        unsigned short l = f32_to_bf16_rne(lo);
        rep_hi[(size_t)row * DDIM + j] = h;
        rep_lo[(size_t)row * DDIM + j] = l;
    }
}

// --------------------------------------------------------------------------
// Kernel 2: fused similarity + exp row-sums. Each wave owns a 16-row strip
// (A kept in VGPRs, hi+lo); block of 8 waves shares double-buffered
// 16-column x 256-K B tiles DMA'd into LDS via global_load_async_to_lds.
// 3x bf16 WMMA per K-step: sim ~= A_hi*B_hi + A_hi*B_lo + A_lo*B_hi.
// --------------------------------------------------------------------------
__global__ __launch_bounds__(256)
void sim_rowsum_kernel(const unsigned short* __restrict__ rep_hi,
                       const unsigned short* __restrict__ rep_lo,
                       float* __restrict__ loss_rows) {
    // Double-buffered B tiles: [buf][16 cols x 32 uint4] per plane = 8 KB each.
    __shared__ uint4 sHi[2][16 * 32];
    __shared__ uint4 sLo[2][16 * 32];

    const int lane    = threadIdx.x & 31;
    const int wave    = threadIdx.x >> 5;
    const int l15     = lane & 15;
    const int laneHi  = lane >> 4;
    const int m0      = (blockIdx.x * 8 + wave) * 16;

    const uint4* gHi = (const uint4*)rep_hi;   // 32 uint4 per 256-col row
    const uint4* gLo = (const uint4*)rep_lo;

    // ---- Preload A strip (rows m0..m0+15, full K) into registers ----------
    // A 16x32 bf16 layout: M = lane&15; lanes 0-15 hold K {0..7,16..23},
    // lanes 16-31 hold K {8..15,24..31} within each 32-wide K step.
    ABFrag a_hi[KSTEPS], a_lo[KSTEPS];
    {
        const int arow = m0 + l15;
        #pragma unroll
        for (int ks = 0; ks < KSTEPS; ++ks) {
            const int i0 = arow * 32 + ks * 4 + laneHi;  // (col = ks*32 + laneHi*8)/8
            const int i1 = i0 + 2;                       // col + 16
            a_hi[ks].u[0] = gHi[i0];  a_hi[ks].u[1] = gHi[i1];
            a_lo[ks].u[0] = gLo[i0];  a_lo[ks].u[1] = gLo[i1];
        }
    }

    // Per-thread DMA slice: 2 x b128 per plane (512 uint4 / 256 threads).
    const int li0  = threadIdx.x;          // 0..255
    const int li1  = threadIdx.x + 256;    // 256..511
    const int row0 = li0 >> 5, c40 = li0 & 31;
    const int row1 = li1 >> 5, c41 = li1 & 31;

    // Issue DMA of column tile nt into buffer b.
    auto issue_tile = [&](int nt, int b) {
        const int n0 = nt * 16;
        async_load_b128((unsigned)(uintptr_t)&sHi[b][li0], gHi + ((n0 + row0) * 32 + c40));
        async_load_b128((unsigned)(uintptr_t)&sHi[b][li1], gHi + ((n0 + row1) * 32 + c41));
        async_load_b128((unsigned)(uintptr_t)&sLo[b][li0], gLo + ((n0 + row0) * 32 + c40));
        async_load_b128((unsigned)(uintptr_t)&sLo[b][li1], gLo + ((n0 + row1) * 32 + c41));
    };

    float rowsum[8], diag_e[8], posv[8];
    #pragma unroll
    for (int r = 0; r < 8; ++r) { rowsum[r] = 0.0f; diag_e[r] = 0.0f; posv[r] = 0.0f; }

    // ---- Pipeline prologue: DMA tile 0 into buffer 0 -----------------------
    issue_tile(0, 0);
    wait_async0();
    __syncthreads();

    for (int nt = 0; nt < NTILES; ++nt) {
        const int cur = nt & 1;
        // Overlap: DMA next tile into the other buffer while computing.
        if (nt + 1 < NTILES) issue_tile(nt + 1, cur ^ 1);

        // ---- 16x16 tile: C = A x B^T via 3-term bf16 split ----------------
        const int n0 = nt * 16;
        v8f c = {};
        #pragma unroll
        for (int ks = 0; ks < KSTEPS; ++ks) {
            // B 32x16 layout: N = lane&15; lanes 0-15 K {0..7,16..23}, lanes 16-31 K {8..15,24..31}.
            const int col = ks * 32 + laneHi * 8;
            const int li  = l15 * 32 + (col >> 3);
            ABFrag b_hi, b_lo;
            b_hi.u[0] = sHi[cur][li];  b_hi.u[1] = sHi[cur][li + 2];
            b_lo.u[0] = sLo[cur][li];  b_lo.u[1] = sLo[cur][li + 2];

            c = __builtin_amdgcn_wmma_f32_16x16x32_bf16(
                    false, a_hi[ks].v, false, b_hi.v, (short)0, c, false, false);
            c = __builtin_amdgcn_wmma_f32_16x16x32_bf16(
                    false, a_hi[ks].v, false, b_lo.v, (short)0, c, false, false);
            c = __builtin_amdgcn_wmma_f32_16x16x32_bf16(
                    false, a_lo[ks].v, false, b_hi.v, (short)0, c, false, false);
        }

        // ---- Epilogue: exp(invT*sim), capture diag + positive -------------
        // C layout: VGPR r -> M = r + 8*laneHi; lane -> N = lane&15.
        const int coln = n0 + l15;
        #pragma unroll
        for (int r = 0; r < 8; ++r) {
            const int rowg = m0 + r + 8 * laneHi;
            const float val = c[r] * INV_T;
            const float e = expf(val);
            rowsum[r] += e;
            if (coln == rowg)              diag_e[r] = e;    // self-similarity term
            if (coln == (rowg ^ BHALF))    posv[r]   = val;  // positive pair (already *invT)
        }

        // Next buffer must be resident in LDS across all waves before reuse.
        wait_async0();
        __syncthreads();
    }

    // ---- Cross-lane reduce within each 16-lane group (rows are per-group) --
    #pragma unroll
    for (int r = 0; r < 8; ++r) {
        float s = rowsum[r], d = diag_e[r], p = posv[r];
        #pragma unroll
        for (int m = 1; m < 16; m <<= 1) {
            s += __shfl_xor(s, m, 32);
            d += __shfl_xor(d, m, 32);
            p += __shfl_xor(p, m, 32);
        }
        if (l15 == 0) {
            const int rowg = m0 + r + 8 * laneHi;
            // loss_i = log(denominator_i) - positive_i * invT
            loss_rows[rowg] = logf(s - d) - p;
        }
    }
}

// --------------------------------------------------------------------------
// Kernel 3: deterministic fixed-order reduction of 8192 row losses.
// --------------------------------------------------------------------------
__global__ __launch_bounds__(256)
void final_reduce_kernel(const float* __restrict__ loss_rows,
                         float* __restrict__ out) {
    __shared__ float sm[256];
    const int t = threadIdx.x;
    float s = 0.0f;
    for (int i = t; i < ROWS2B; i += 256) s += loss_rows[i];
    sm[t] = s;
    __syncthreads();
    for (int w = 128; w > 0; w >>= 1) {
        if (t < w) sm[t] += sm[t + w];
        __syncthreads();
    }
    if (t == 0) out[0] = sm[0] / (float)ROWS2B;
}

// --------------------------------------------------------------------------
extern "C" void kernel_launch(void* const* d_in, const int* in_sizes, int n_in,
                              void* d_out, int out_size, void* d_ws, size_t ws_size,
                              hipStream_t stream) {
    (void)in_sizes; (void)n_in; (void)out_size; (void)ws_size;
    const float* emb_i = (const float*)d_in[0];
    const float* emb_j = (const float*)d_in[1];
    float* out = (float*)d_out;

    // Workspace layout:
    //   [0,            4 MB)  rep_hi  : 8192*256 bf16 (as u16)
    //   [4 MB,         8 MB)  rep_lo  : 8192*256 bf16 (as u16)
    //   [8 MB,  8 MB + 32KB)  loss_rows: 8192 f32
    char* ws = (char*)d_ws;
    const size_t planeBytes = (size_t)ROWS2B * DDIM * sizeof(unsigned short);
    unsigned short* rep_hi = (unsigned short*)(ws);
    unsigned short* rep_lo = (unsigned short*)(ws + planeBytes);
    float* loss_rows       = (float*)(ws + 2 * planeBytes);

    // 1) normalize + bf16 hi/lo split: 8192 rows, 8 rows (waves) per block
    normalize_split_kernel<<<ROWS2B / 8, 256, 0, stream>>>(emb_i, emb_j, rep_hi, rep_lo);

    // 2) fused WMMA similarity + exp rowsums: 512 row strips / 8 waves = 64 blocks
    sim_rowsum_kernel<<<(ROWS2B / 16) / 8, 256, 0, stream>>>(rep_hi, rep_lo, loss_rows);

    // 3) deterministic scalar reduction
    final_reduce_kernel<<<1, 256, 0, stream>>>(loss_rows, out);
}
